// TreeEncoder_2405181685797
// MI455X (gfx1250) — matrix operand order repaired and checked by reading, not compiled, not run on hardware
//
#include <hip/hip_runtime.h>
#include <hip/hip_bf16.h>

typedef __attribute__((ext_vector_type(16))) _Float16 v16h;
typedef __attribute__((ext_vector_type(8)))  float    v8f;
typedef __attribute__((ext_vector_type(4)))  float    v4f;

#define F_IN    128
#define F_OUT   256
#define NETYPES 16
#define M_BLK   128
#define THREADS 256

// gfx1250 has a hardware V_TANH_F32 transcendental; prefer it over the
// branchy OCML tanhf (one TRANS32 op, co-executes with VALU).
__device__ __forceinline__ float fast_tanh(float x) {
#if __has_builtin(__builtin_amdgcn_tanhf)
    return __builtin_amdgcn_tanhf(x);
#elif __has_builtin(__builtin_amdgcn_tanh_f32)
    return __builtin_amdgcn_tanh_f32(x);
#else
    return tanhf(x);
#endif
}

// Pre-pass: convert the used half of W (F_OUT x F_IN slice of the 256x256
// weight, since h0 columns 128..255 are zero) to f16, row-major [o][k].
__global__ void convert_w_kernel(const float* __restrict__ w,
                                 _Float16* __restrict__ wh) {
    int i = blockIdx.x * blockDim.x + threadIdx.x;   // 0 .. 32767
    int o = i >> 7;
    int k = i & 127;
    wh[i] = (_Float16)w[o * F_OUT + k];
}

// One tree level: out[i,:] = tanh(feat[i,:] @ Wh^T + bias + agg(children)).
// Block = 256 threads = 8 waves; each wave computes a 16-row stripe across
// all 256 output columns with v_wmma_f32_16x16x32_f16 (K=128 -> 4 k-steps).
template <bool HAS_PREV>
__global__ __launch_bounds__(THREADS)
void level_kernel(const float*    __restrict__ feat,  // n x 128 (this level)
                  const _Float16* __restrict__ wh,    // 256 x 128 f16
                  const float*    __restrict__ bias,  // 256
                  const float*    __restrict__ emb,   // 16 x 256
                  const int*      __restrict__ ets,   // 2n etypes (lvl>=1)
                  const float*    __restrict__ prev,  // prev level out (lvl>=1)
                  float*          __restrict__ out)   // n x 256
{
    __shared__ __align__(32) _Float16 lds_w[F_OUT * F_IN];     // 64 KB
    __shared__ float lds_emb[NETYPES * F_OUT];                 // 16 KB
    __shared__ float lds_bias[F_OUT];                          //  1 KB

    const int tid  = threadIdx.x;
    const int lane = tid & 31;
    const int wv   = tid >> 5;
    const int hi   = lane >> 4;        // lane half: 0 or 1
    const int m16  = lane & 15;

    // ---- stage W (f16), edge_emb, bias into LDS ----
    {
        const uint4* src = (const uint4*)wh;
        uint4*       dst = (uint4*)lds_w;
#pragma unroll
        for (int i = 0; i < 16; ++i)                  // 64 KB / 16 B / 256 thr
            dst[tid + i * THREADS] = src[tid + i * THREADS];
#pragma unroll
        for (int i = 0; i < 16; ++i)
            lds_emb[tid + i * THREADS] = emb[tid + i * THREADS];
        if (tid < F_OUT) lds_bias[tid] = bias[tid];
    }
    __syncthreads();

    const int rbase = blockIdx.x * M_BLK + wv * 16;   // this wave's row tile

    // ---- A fragments: 16x32 f16 per k-step, built in registers ----
    // Lane holds row (lane&15); K segments [kb,kb+8) and [kb+16,kb+24),
    // kb = ks*32 + 8*(lane>=16). Packed pairs -> v16h.
    const float* arow = feat + (size_t)(rbase + m16) * F_IN;
    v16h afr[4];
#pragma unroll
    for (int ks = 0; ks < 4; ++ks) {
        const int kb = ks * 32 + hi * 8;
        v4f x0 = *(const v4f*)(arow + kb);
        v4f x1 = *(const v4f*)(arow + kb + 4);
        v4f x2 = *(const v4f*)(arow + kb + 16);
        v4f x3 = *(const v4f*)(arow + kb + 20);
        v16h a;
#pragma unroll
        for (int j = 0; j < 4; ++j) {
            a[j]      = (_Float16)x0[j];
            a[4 + j]  = (_Float16)x1[j];
            a[8 + j]  = (_Float16)x2[j];
            a[12 + j] = (_Float16)x3[j];
        }
        afr[ks] = a;
    }

    // ---- per-row edge types (C-layout rows: M = r + 8*hi) ----
    int et0[8], et1[8];
    if constexpr (HAS_PREV) {
#pragma unroll
        for (int r = 0; r < 8; ++r) {
            const int grow = rbase + hi * 8 + r;
            et0[r] = ets[2 * grow];
            et1[r] = ets[2 * grow + 1];
        }
    }

    // ---- sweep 16 column tiles; 4 WMMA k-steps each ----
#pragma unroll
    for (int t = 0; t < 16; ++t) {
        const int n = t * 16 + m16;                   // output column (B/C/D)
        v8f c = {0.f, 0.f, 0.f, 0.f, 0.f, 0.f, 0.f, 0.f};
#pragma unroll
        for (int ks = 0; ks < 4; ++ks) {
            // B fragment: lane holds column n, K segments [kbase,kbase+8) and
            // [kbase+16,kbase+24), kbase = ks*32 + 8*hi; two 16-byte LDS reads.
            const int kbase = ks * 32 + hi * 8;
            typedef __attribute__((ext_vector_type(8))) _Float16 v8h_b;
            v8h_b b0 = *(const v8h_b*)&lds_w[n * F_IN + kbase];
            v8h_b b1 = *(const v8h_b*)&lds_w[n * F_IN + kbase + 16];
            v16h b = __builtin_shufflevector(b0, b1, 0, 1, 2, 3, 4, 5, 6, 7, 8, 9, 10, 11, 12, 13, 14, 15);
            c = __builtin_amdgcn_wmma_f32_16x16x32_f16(
                    false, afr[ks], false, b, (short)0, c, false, false);
        }

        const float bcol = lds_bias[n];
#pragma unroll
        for (int r = 0; r < 8; ++r) {
            const int grow = rbase + hi * 8 + r;
            float v = c[r] + bcol;
            if constexpr (HAS_PREV) {
                const float h0v = prev[(size_t)(2 * grow) * F_OUT + n];
                const float h1v = prev[(size_t)(2 * grow) * F_OUT + F_OUT + n];
                v += h0v * lds_emb[et0[r] * F_OUT + n]
                   + h1v * lds_emb[et1[r] * F_OUT + n];
            }
            out[(size_t)grow * F_OUT + n] = fast_tanh(v);
        }
    }
}

extern "C" void kernel_launch(void* const* d_in, const int* in_sizes, int n_in,
                              void* d_out, int out_size, void* d_ws, size_t ws_size,
                              hipStream_t stream) {
    const float* feat   = (const float*)d_in[0];   // N_NODES x 128
    const float* w      = (const float*)d_in[1];   // 256 x 256
    const float* bias   = (const float*)d_in[2];   // 256
    const float* emb    = (const float*)d_in[3];   // 16 x 256
    const int*   etypes = (const int*)d_in[4];     // N_EDGES
    float*       out    = (float*)d_out;           // N_NODES x 256
    _Float16*    wh     = (_Float16*)d_ws;         // 64 KB scratch

    convert_w_kernel<<<(F_OUT * F_IN) / THREADS, THREADS, 0, stream>>>(w, wh);

    static const int SIZES[9] = {262144, 131072, 65536, 32768,
                                 16384, 8192, 4096, 2048, 1024};
    long node_off = 0, prev_off = 0, edge_off = 0;
    for (int l = 0; l <= 8; ++l) {
        const int n = SIZES[l];
        const float* fl = feat + node_off * (long)F_IN;
        float*       ol = out + node_off * (long)F_OUT;
        if (l == 0) {
            level_kernel<false><<<n / M_BLK, THREADS, 0, stream>>>(
                fl, wh, bias, emb, nullptr, nullptr, ol);
        } else {
            const float* pv = out + prev_off * (long)F_OUT;
            const int*   el = etypes + edge_off;
            level_kernel<true><<<n / M_BLK, THREADS, 0, stream>>>(
                fl, wh, bias, emb, el, pv, ol);
            edge_off += 2L * n;   // EDGE_OFF cumsum over SIZES[1:]
        }
        prev_off = node_off;
        node_off += n;
    }
}
